// GCN_3702261809343
// MI455X (gfx1250) — matrix operand order repaired and checked by reading, not compiled, not run on hardware
//
#include <hip/hip_runtime.h>
#include <hip/hip_bf16.h>
#include <stdint.h>

// ---------------------------------------------------------------------------
// GCN forward on MI455X (gfx1250, wave32, WMMA).
//   h  = leaky( D^-1/2 (A+I) D^-1/2 (x @ Wgc) + bgc )   [131072,128]
//   y1 = leaky( h.reshape(1024,16384) @ W1 + b1 )        [1024,512]
//   out= y1 @ W2 + b2                                    [1024,128]
// GEMMs: v_wmma_f32_16x16x32_bf16. B staged via global_load_async_to_lds_b128
// (double buffered, shared by 8 waves/block). Each wave does 32x64 output
// (2 M-tiles): every B fragment feeds 2 WMMAs. Big FC GEMMs use split-K with
// f32 atomic partials for occupancy.
// ---------------------------------------------------------------------------

typedef __attribute__((ext_vector_type(16))) __bf16 v16bf;
typedef __attribute__((ext_vector_type(8)))  __bf16 v8bf;
typedef __attribute__((ext_vector_type(8)))  float  v8f;
typedef __attribute__((ext_vector_type(4)))  int    v4i;

typedef __attribute__((address_space(1))) v4i* gv4ip;   // global v4i*
typedef __attribute__((address_space(3))) v4i* sv4ip;   // LDS v4i*

#define N_NODES  131072
#define FEAT     128
#define HID      128
#define NPG      128          // nodes per graph
#define NGRAPH   (N_NODES / NPG)
#define FC_HID   512
#define ACT_DIM  128
#define N_EDGES  2097152

#if defined(__has_builtin)
#  if __has_builtin(__builtin_amdgcn_global_load_async_to_lds_b128)
#    define USE_ASYNC_LDS 1
#  endif
#  if __has_builtin(__builtin_amdgcn_s_wait_asynccnt)
#    define HAVE_WAIT_ASYNC_BUILTIN 1
#  endif
#endif

__device__ __forceinline__ float leaky(float v) { return v >= 0.f ? v : 0.01f * v; }

// 16B-per-thread stage of packed B into LDS (async when toolchain supports it)
__device__ __forceinline__ void stage16(const __bf16* __restrict__ src, __bf16* dst) {
#if USE_ASYNC_LDS
    __builtin_amdgcn_global_load_async_to_lds_b128((gv4ip)src, (sv4ip)dst, 0, 0);
#else
    *(v8bf*)dst = *(const v8bf*)src;     // global_load_b128 + ds_store_b128
#endif
}

__device__ __forceinline__ void stage_wait() {
#if USE_ASYNC_LDS
#  if HAVE_WAIT_ASYNC_BUILTIN
    __builtin_amdgcn_s_wait_asynccnt(0);
#  else
    asm volatile("s_wait_asynccnt 0x0" ::: "memory");
#  endif
#endif
}

// ---------------------------------------------------------------------------
// Elementwise / graph kernels
// ---------------------------------------------------------------------------
__global__ void zero_f32_kernel(float* p, int n) {
    int t = blockIdx.x * blockDim.x + threadIdx.x;
    if (t < n) p[t] = 0.0f;
}

__global__ void deg_init_kernel(float* deg, int n) {
    int t = blockIdx.x * blockDim.x + threadIdx.x;
    if (t < n) deg[t] = 1.0f;                 // self-loop contributes 1
}

__global__ void deg_edges_kernel(const int* __restrict__ dst, float* deg, int nE) {
    int t = blockIdx.x * blockDim.x + threadIdx.x;
    if (t < nE) atomicAdd(&deg[dst[t]], 1.0f);
}

__global__ void rsqrt_deg_kernel(const float* __restrict__ deg, float* dinv, int n) {
    int t = blockIdx.x * blockDim.x + threadIdx.x;
    if (t < n) { float d = deg[t]; dinv[t] = d > 0.f ? rsqrtf(d) : 0.f; }
}

__global__ void f32_to_bf16_kernel(const float* __restrict__ in, __bf16* out, int n) {
    int t = blockIdx.x * blockDim.x + threadIdx.x;
    if (t < n) out[t] = (__bf16)in[t];
}

// xs = xw * dinv[node]  (in place over xw); acc initialized with self-loop term xs
__global__ void xs_init_kernel(float* xw, float* acc, const float* __restrict__ dinv, int total) {
    int t = blockIdx.x * blockDim.x + threadIdx.x;
    if (t < total) {
        float v = xw[t] * dinv[t >> 7];
        xw[t]  = v;
        acc[t] = v;
    }
}

// one wave per edge; each lane handles 4 consecutive features (float4 load, 4 f32 atomics)
__global__ void scatter_edges_kernel(const int* __restrict__ ei, const float* __restrict__ xs,
                                     float* acc, int nE) {
    size_t tid = (size_t)blockIdx.x * blockDim.x + threadIdx.x;
    int e    = (int)(tid >> 5);
    int lane = (int)(tid & 31);
    if (e >= nE) return;
    int src = ei[e];
    int dst = ei[nE + e];
    const float4 v = *(const float4*)(xs + (size_t)src * HID + lane * 4);
    float* p = acc + (size_t)dst * HID + lane * 4;
    atomicAdd(p + 0, v.x);
    atomicAdd(p + 1, v.y);
    atomicAdd(p + 2, v.z);
    atomicAdd(p + 3, v.w);
}

// h = leaky(dinv[node] * acc + b_gc[feat]) -> bf16 activation buffer for FC1
__global__ void h_finalize_kernel(const float* __restrict__ acc, const float* __restrict__ dinv,
                                  const float* __restrict__ bgc, __bf16* hb, int total) {
    int t = blockIdx.x * blockDim.x + threadIdx.x;
    if (t < total) {
        float v = dinv[t >> 7] * acc[t] + bgc[t & (HID - 1)];
        hb[t] = (__bf16)leaky(v);
    }
}

// split-K finalizers
__global__ void bias_act_bf16_kernel(const float* __restrict__ Cf, const float* __restrict__ bias,
                                     __bf16* outb, int N, int total) {
    int t = blockIdx.x * blockDim.x + threadIdx.x;
    if (t < total) outb[t] = (__bf16)leaky(Cf[t] + bias[t % N]);
}

__global__ void bias_inplace_kernel(float* C, const float* __restrict__ bias, int N, int total) {
    int t = blockIdx.x * blockDim.x + threadIdx.x;
    if (t < total) C[t] = C[t] + bias[t % N];
}

// ---------------------------------------------------------------------------
// Pack row-major fp32 weight W[K,N] into WMMA B-fragment order (bf16).
// Tile = 32(K) x 16(N): element (lane, j) <- W[kt*32 + (lane>>4)*16 + j, nt*16 + (lane&15)]
// so each lane's 16 values are contiguous 32 bytes at load time, and the 4
// tiles of a 64-col strip are a contiguous 4KB chunk (one block-stage).
// ---------------------------------------------------------------------------
__global__ void pack_b_kernel(const float* __restrict__ W, __bf16* Wp, int K, int N) {
    int t = blockIdx.x * blockDim.x + threadIdx.x;
    if (t >= K * N) return;
    int tile   = t >> 9;          // 512 elems per 32x16 tile
    int within = t & 511;
    int lane   = within >> 4;
    int j      = within & 15;
    int nTiles = N >> 4;
    int kt = tile / nTiles;
    int nt = tile - kt * nTiles;
    int n  = (nt << 4) + (lane & 15);
    int k  = (kt << 5) + ((lane >> 4) << 4) + j;
    Wp[t] = (__bf16)W[(size_t)k * N + n];
}

// ---------------------------------------------------------------------------
// WMMA GEMM: C[M,N] = act(A[M,K](bf16,row-major) @ Bpacked + bias)
// Block = 256 thr = 8 waves covering 256(M) x 64(N); each wave owns 32x64
// (2 M-tiles x 4 N-tiles => 8 WMMAs per K-step, each B frag used twice).
// B strip staged in LDS (double buffered, async prefetch overlaps WMMAs).
// SPLITK: gridDim.y K-slices accumulate into f32 C via atomics (bias/act
// deferred to a finalize pass). Requires M%256==0, N%64==0, K%(32*gridDim.y)==0.
// ---------------------------------------------------------------------------
template<bool OUT_BF16, bool SPLITK>
__global__ void __launch_bounds__(256)
gemm_bf16_wmma(const __bf16* __restrict__ A, const __bf16* __restrict__ Bp,
               const float* __restrict__ bias, void* __restrict__ Cout,
               int M, int N, int K, int act) {
    __shared__ __align__(16) __bf16 sB[2][2048];   // 2 x 4KB: 4 B-tiles per buffer

    const int tid  = threadIdx.x;
    const int lane = tid & 31;
    const int wave = tid >> 5;

    const int nStrips = N >> 6;
    const int mBlock  = blockIdx.x / nStrips;
    const int strip   = blockIdx.x - mBlock * nStrips;
    const int mBase = mBlock * 256 + wave * 32;    // wave's 32 rows
    const int n0 = strip << 6;
    const int nTiles = N >> 4;

    // K slice for this block (split-K)
    const int kPer  = K / gridDim.y;
    const int ktBeg = (blockIdx.y * kPer) >> 5;
    const int ktEnd = ktBeg + (kPer >> 5);

    // A fragment addressing (16x32 bf16 layout): lane<16 -> row=lane, K in {0..15}
    const int row = lane & 15;
    const int kb  = (lane >> 4) << 4;
    const __bf16* Arow0 = A + (size_t)(mBase + row) * K + kb;
    const __bf16* Arow1 = Arow0 + (size_t)16 * K;

    // Packed-B strip for this block: 4KB contiguous per K-step
    const __bf16* Bstrip = Bp + (size_t)(n0 >> 4) * 512 + (size_t)tid * 8;
    const size_t  Bstep  = (size_t)nTiles * 512;   // elems per K-step

    v8f acc00 = {}, acc01 = {}, acc02 = {}, acc03 = {};
    v8f acc10 = {}, acc11 = {}, acc12 = {}, acc13 = {};

    stage16(Bstrip + (size_t)ktBeg * Bstep, &sB[0][tid * 8]);   // prologue
    int p = 0;
    for (int kt = ktBeg; kt < ktEnd; ++kt) {
        stage_wait();
        __syncthreads();                           // sB[p] ready for everyone
        if (kt + 1 < ktEnd)                        // prefetch next step into other buffer
            stage16(Bstrip + (size_t)(kt + 1) * Bstep, &sB[p ^ 1][tid * 8]);

        const int ko = kt << 5;
        v8bf lo0 = *(const v8bf*)(Arow0 + ko);
        v8bf hi0 = *(const v8bf*)(Arow0 + ko + 8);
        v8bf lo1 = *(const v8bf*)(Arow1 + ko);
        v8bf hi1 = *(const v8bf*)(Arow1 + ko + 8);
        v16bf a0 = __builtin_shufflevector(lo0, hi0, 0,1,2,3,4,5,6,7,8,9,10,11,12,13,14,15);
        v16bf a1 = __builtin_shufflevector(lo1, hi1, 0,1,2,3,4,5,6,7,8,9,10,11,12,13,14,15);

        const __bf16* bt = &sB[p][lane * 16];
        v16bf b0 = *(const v16bf*)(bt);
        v16bf b1 = *(const v16bf*)(bt + 512);
        v16bf b2 = *(const v16bf*)(bt + 1024);
        v16bf b3 = *(const v16bf*)(bt + 1536);

        acc00 = __builtin_amdgcn_wmma_f32_16x16x32_bf16(false, a0, false, b0, (short)0, acc00, false, false);
        acc10 = __builtin_amdgcn_wmma_f32_16x16x32_bf16(false, a1, false, b0, (short)0, acc10, false, false);
        acc01 = __builtin_amdgcn_wmma_f32_16x16x32_bf16(false, a0, false, b1, (short)0, acc01, false, false);
        acc11 = __builtin_amdgcn_wmma_f32_16x16x32_bf16(false, a1, false, b1, (short)0, acc11, false, false);
        acc02 = __builtin_amdgcn_wmma_f32_16x16x32_bf16(false, a0, false, b2, (short)0, acc02, false, false);
        acc12 = __builtin_amdgcn_wmma_f32_16x16x32_bf16(false, a1, false, b2, (short)0, acc12, false, false);
        acc03 = __builtin_amdgcn_wmma_f32_16x16x32_bf16(false, a0, false, b3, (short)0, acc03, false, false);
        acc13 = __builtin_amdgcn_wmma_f32_16x16x32_bf16(false, a1, false, b3, (short)0, acc13, false, false);

        __syncthreads();                           // all reads of sB[p] done before restage
        p ^= 1;
    }

    // C/D layout: VGPR r, lanes 0-15 -> (M=r, N=lane); lanes 16-31 -> (M=8+r, N=lane-16)
    const int cLane = lane & 15;
    v8f accs[2][4] = {{acc00, acc01, acc02, acc03}, {acc10, acc11, acc12, acc13}};
#pragma unroll
    for (int i = 0; i < 2; ++i) {
        const int rBase = mBase + i * 16 + ((lane >> 4) << 3);
#pragma unroll
        for (int t = 0; t < 4; ++t) {
            const int col = n0 + (t << 4) + cLane;
            const float bv = (!SPLITK && bias) ? bias[col] : 0.f;
#pragma unroll
            for (int r = 0; r < 8; ++r) {
                float v = accs[i][t][r] + bv;
                size_t idx = (size_t)(rBase + r) * N + col;
                if (SPLITK) {
                    atomicAdd((float*)Cout + idx, v);
                } else {
                    if (act) v = leaky(v);
                    if (OUT_BF16) ((__bf16*)Cout)[idx] = (__bf16)v;
                    else          ((float*)Cout)[idx]  = v;
                }
            }
        }
    }
}

// ---------------------------------------------------------------------------
// Host-side orchestration
// ---------------------------------------------------------------------------
static inline int ceil_div(long long a, int b) { return (int)((a + b - 1) / b); }

extern "C" void kernel_launch(void* const* d_in, const int* in_sizes, int n_in,
                              void* d_out, int out_size, void* d_ws, size_t ws_size,
                              hipStream_t stream) {
    const float* x   = (const float*)d_in[0];
    const int*   ei  = (const int*)  d_in[1];   // [2, N_EDGES]: row0=src, row1=dst
    const float* Wgc = (const float*)d_in[2];
    const float* bgc = (const float*)d_in[3];
    const float* W1  = (const float*)d_in[4];
    const float* b1  = (const float*)d_in[5];
    const float* W2  = (const float*)d_in[6];
    const float* b2  = (const float*)d_in[7];
    float* out = (float*)d_out;

    // Workspace carve-up (256B aligned)
    char* ws = (char*)d_ws;
    size_t off = 0;
    auto carve = [&](size_t bytes) -> char* {
        off = (off + 255) & ~(size_t)255;
        char* p = ws + off;
        off += bytes;
        return p;
    };
    float*  deg  = (float*) carve((size_t)N_NODES * 4);
    float*  dinv = (float*) carve((size_t)N_NODES * 4);
    __bf16* xb   = (__bf16*)carve((size_t)N_NODES * FEAT * 2);   // x in bf16; reused as h (bf16)
    __bf16* WgcP = (__bf16*)carve((size_t)FEAT * HID * 2);
    float*  xw   = (float*) carve((size_t)N_NODES * HID * 4);    // xw, then xs (in place)
    float*  acc  = (float*) carve((size_t)N_NODES * HID * 4);
    __bf16* W1P  = (__bf16*)carve((size_t)NPG * HID * FC_HID * 2);
    float*  y1f  = (float*) carve((size_t)NGRAPH * FC_HID * 4);  // split-K partials
    __bf16* y1   = (__bf16*)carve((size_t)NGRAPH * FC_HID * 2);
    __bf16* W2P  = (__bf16*)carve((size_t)FC_HID * ACT_DIM * 2);

    const int B = 256;

    // 1) degrees (with self loops) and D^-1/2
    deg_init_kernel  <<<ceil_div(N_NODES, B), B, 0, stream>>>(deg, N_NODES);
    deg_edges_kernel <<<ceil_div(N_EDGES, B), B, 0, stream>>>(ei + N_EDGES, deg, N_EDGES);
    rsqrt_deg_kernel <<<ceil_div(N_NODES, B), B, 0, stream>>>(deg, dinv, N_NODES);

    // 2) bf16 conversions / weight packing
    f32_to_bf16_kernel<<<ceil_div((long long)N_NODES * FEAT, B), B, 0, stream>>>(x, xb, N_NODES * FEAT);
    pack_b_kernel<<<ceil_div(FEAT * HID, B),                 B, 0, stream>>>(Wgc, WgcP, FEAT, HID);
    pack_b_kernel<<<ceil_div((long long)NPG*HID*FC_HID, B),  B, 0, stream>>>(W1,  W1P,  NPG * HID, FC_HID);
    pack_b_kernel<<<ceil_div(FC_HID * ACT_DIM, B),           B, 0, stream>>>(W2,  W2P,  FC_HID, ACT_DIM);

    // 3) xw = x @ Wgc  (WMMA, direct store): 512 mBlocks x 2 strips
    gemm_bf16_wmma<false, false><<<dim3((N_NODES >> 8) * (HID >> 6), 1), 256, 0, stream>>>(
        xb, WgcP, nullptr, xw, N_NODES, HID, FEAT, 0);

    // 4) xs = xw * dinv (in place); acc = xs (self-loop term)
    xs_init_kernel<<<ceil_div((long long)N_NODES * HID, B), B, 0, stream>>>(xw, acc, dinv, N_NODES * HID);

    // 5) edge scatter: acc[dst] += xs[src]   (one wave per edge, f32 atomics hitting L2)
    scatter_edges_kernel<<<ceil_div((long long)N_EDGES * 32, B), B, 0, stream>>>(ei, xw, acc, N_EDGES);

    // 6) h = leaky(dinv*acc + bgc) -> bf16 (reuse xb as h)
    h_finalize_kernel<<<ceil_div((long long)N_NODES * HID, B), B, 0, stream>>>(acc, dinv, bgc, xb, N_NODES * HID);

    // 7) y1f = h.reshape(1024,16384) @ W1  (WMMA split-K=8 -> 256 blocks), then bias+leaky -> bf16
    zero_f32_kernel<<<ceil_div(NGRAPH * FC_HID, B), B, 0, stream>>>(y1f, NGRAPH * FC_HID);
    gemm_bf16_wmma<false, true><<<dim3((NGRAPH >> 8) * (FC_HID >> 6), 8), 256, 0, stream>>>(
        xb, W1P, nullptr, y1f, NGRAPH, FC_HID, NPG * HID, 0);
    bias_act_bf16_kernel<<<ceil_div(NGRAPH * FC_HID, B), B, 0, stream>>>(
        y1f, b1, y1, FC_HID, NGRAPH * FC_HID);

    // 8) out = y1 @ W2 + b2  (WMMA split-K=4 -> 32 blocks), bias added in place
    zero_f32_kernel<<<ceil_div(NGRAPH * ACT_DIM, B), B, 0, stream>>>(out, NGRAPH * ACT_DIM);
    gemm_bf16_wmma<false, true><<<dim3((NGRAPH >> 8) * (ACT_DIM >> 6), 4), 256, 0, stream>>>(
        y1, W2P, nullptr, out, NGRAPH, ACT_DIM, FC_HID, 0);
    bias_inplace_kernel<<<ceil_div(NGRAPH * ACT_DIM, B), B, 0, stream>>>(
        out, b2, ACT_DIM, NGRAPH * ACT_DIM);

    (void)in_sizes; (void)n_in; (void)out_size; (void)ws_size;
}
